// Binarized_CONV_30391188587232
// MI455X (gfx1250) — matrix operand order repaired and checked by reading, not compiled, not run on hardware
//
#include <hip/hip_runtime.h>
#include <math.h>

// CDNA5 (gfx1250) wave32 WMMA types
typedef __attribute__((ext_vector_type(16))) _Float16 v16h;
typedef __attribute__((ext_vector_type(8)))  float    v8f;

#define TB 256

// ---------------------------------------------------------------------------
// Elementwise helpers
// ---------------------------------------------------------------------------
__global__ void k_sign_f16(const float* __restrict__ src, _Float16* __restrict__ dst, int n) {
  int i = blockIdx.x * blockDim.x + threadIdx.x;
  if (i < n) dst[i] = (_Float16)(src[i] >= 0.f ? 1.f : -1.f);
}

__global__ void k_f32_to_f16(const float* __restrict__ src, _Float16* __restrict__ dst, int n) {
  int i = blockIdx.x * blockDim.x + threadIdx.x;
  if (i < n) dst[i] = (_Float16)src[i];
}

__global__ void k_relu_f16(const float* __restrict__ src, _Float16* __restrict__ dst, int n) {
  int i = blockIdx.x * blockDim.x + threadIdx.x;
  if (i < n) dst[i] = (_Float16)fmaxf(src[i], 0.f);
}

__global__ void k_sigmoid(const float* __restrict__ src, float* __restrict__ dst, int n) {
  int i = blockIdx.x * blockDim.x + threadIdx.x;
  if (i < n) dst[i] = 1.f / (1.f + __expf(-src[i]));
}

__global__ void k_maxpool2_f16(const _Float16* __restrict__ in, _Float16* __restrict__ out,
                               int C, int Hi, int Wi, int total /* B*C*Ho*Wo */) {
  int Ho = Hi >> 1, Wo = Wi >> 1;
  int i = blockIdx.x * blockDim.x + threadIdx.x;
  if (i >= total) return;
  int wo = i % Wo; int tmp = i / Wo;
  int ho = tmp % Ho; tmp /= Ho;
  int c = tmp % C; int b = tmp / C;
  const _Float16* p = in + (((b * C + c) * Hi + ho * 2) * Wi + wo * 2);
  float m = fmaxf(fmaxf((float)p[0], (float)p[1]),
                  fmaxf((float)p[Wi], (float)p[Wi + 1]));
  out[i] = (_Float16)m;
}

// ---------------------------------------------------------------------------
// BatchNorm (training mode, biased var): two-pass
// ---------------------------------------------------------------------------
__global__ __launch_bounds__(TB) void k_bn_stats(const float* __restrict__ Y,
                                                 float* __restrict__ mean,
                                                 float* __restrict__ invstd,
                                                 int C, int S, int Btot) {
  int c = blockIdx.x;
  int t = threadIdx.x;
  int per = Btot * S;
  float s = 0.f, s2 = 0.f;
  for (int i = t; i < per; i += TB) {
    int b = i / S, sp = i - b * S;
    float v = Y[((long)b * C + c) * S + sp];
    s += v; s2 += v * v;
  }
  __shared__ float rs[TB], rs2[TB];
  rs[t] = s; rs2[t] = s2;
  __syncthreads();
  for (int off = TB / 2; off > 0; off >>= 1) {
    if (t < off) { rs[t] += rs[t + off]; rs2[t] += rs2[t + off]; }
    __syncthreads();
  }
  if (t == 0) {
    float m = rs[0] / (float)per;
    float var = rs2[0] / (float)per - m * m;
    mean[c] = m;
    invstd[c] = rsqrtf(var + 1e-5f);
  }
}

__global__ void k_bn_apply_relu(const float* __restrict__ Y, _Float16* __restrict__ out,
                                const float* __restrict__ mean, const float* __restrict__ invstd,
                                const float* __restrict__ g, const float* __restrict__ bias,
                                int C, int S, int total) {
  int i = blockIdx.x * blockDim.x + threadIdx.x;
  if (i >= total) return;
  int c = (i / S) % C;
  float v = (Y[i] - mean[c]) * invstd[c] * g[c] + bias[c];
  out[i] = (_Float16)fmaxf(v, 0.f);
}

// ---------------------------------------------------------------------------
// Implicit-GEMM conv via WMMA f16->f32, shape-templated (constant divisors).
//   D[M = Cout, N = B*Ho*Wo] = Wb[M, K] x im2col[K, N],   K = Cin*KH*KW
//
// Block = 256 threads = 8 waves (4x2 wave grid). Block tile 128(M) x 64(N),
// K-tile 64 per LDS stage. Each wave owns a 32x32 accumulator block
// (4 x v8f) and issues 4 WMMAs per 32-wide K-substep => 8 WMMAs per stage.
//
// LDS is staged in *fragment order*: As[ks][mtile][lane][16 halves],
// Bs[ks][ntile][lane][16 halves], so a wave loads each fragment as one
// aligned 32B read (2 x ds_load_b128) instead of 16 scalar gathers.
// The ISA 7.12.2 half->K permutation is applied at staging time:
//   A (16x32): lanes 0-15 M=lane, h0-7 -> K0-7, h8-15 -> K16-23;
//              lanes 16-31 shift K by +8.
//   B (32x16): N = lane&15; same K permutation as A.
// EXEC is all-ones around the WMMAs (bounds handled by zero-padding LDS).
// Output layout: Y[b, m, sp] = [B, M, Ho*Wo]  (== [B, M] for FC, S==1).
// ---------------------------------------------------------------------------
template<int CIN, int H, int W, int KH, int KW>
__global__ __launch_bounds__(TB) void k_conv_wmma(
    const _Float16* __restrict__ Wb,   // [M, K] binarized weights (+-1)
    const _Float16* __restrict__ X,    // [B, CIN, H, W]
    float* __restrict__ Y,             // [B, M, Ho, Wo]
    int M, int Btot)
{
  constexpr int Ho  = H - KH + 1;
  constexpr int Wo  = W - KW + 1;
  constexpr int S   = Ho * Wo;
  constexpr int KHW = KH * KW;
  constexpr int K   = CIN * KHW;
  constexpr int BM = 128, BN = 64, BK = 64;
  const int N = Btot * S;

  // fragment-order LDS: [ksub(2)][subtile][lane(32)][half(16)]
  __shared__ __attribute__((aligned(32))) _Float16 As[2 * 8 * 32 * 16]; // 16 KB
  __shared__ __attribute__((aligned(32))) _Float16 Bs[2 * 4 * 32 * 16]; //  8 KB

  const int t    = threadIdx.x;
  const int lane = t & 31;
  const int wave = t >> 5;
  const int wm   = wave & 3;   // wave M block: rows wm*32 (m-subtiles 2wm, 2wm+1)
  const int wn   = wave >> 2;  // wave N block: cols wn*32 (n-subtiles 2wn, 2wn+1)
  const int hi16 = lane >> 4;
  const int l15  = lane & 15;

  const int m0 = blockIdx.y * BM;
  const int n0 = blockIdx.x * BN;

  v8f acc[2][2] = {};

  for (int k0 = 0; k0 < K; k0 += BK) {
    // ---- stage A (128 x 64) in fragment order ----
    #pragma unroll
    for (int i = t; i < 2 * 8 * 32 * 16; i += TB) {
      int h  = i & 15;
      int ln = (i >> 4) & 31;
      int mt = (i >> 9) & 7;
      int ks = i >> 12;
      int kk = (h & 7) + ((h >> 3) << 4) + ((ln >> 4) << 3) + (ks << 5);
      int m  = m0 + (mt << 4) + (ln & 15);
      int k  = k0 + kk;
      _Float16 v = (_Float16)0.f;
      if (m < M && k < K) v = Wb[m * K + k];
      As[i] = v;
    }
    // ---- stage B (64 x 64) in fragment order, implicit im2col ----
    #pragma unroll
    for (int i = t; i < 2 * 4 * 32 * 16; i += TB) {
      int h  = i & 15;
      int ln = (i >> 4) & 31;
      int nt = (i >> 9) & 3;
      int ks = i >> 11;
      int kk = (h & 7) + ((h >> 3) << 4) + ((ln >> 4) << 3) + (ks << 5);
      int n  = n0 + (nt << 4) + (ln & 15);
      int k  = k0 + kk;
      _Float16 v = (_Float16)0.f;
      if (k < K && n < N) {
        int cin = k / KHW;  int r  = k - cin * KHW;   // constant divisors
        int kh  = r / KW;   int kw = r - kh * KW;
        int b   = n / S;    int sp = n - b * S;
        int ho  = sp / Wo;  int wo = sp - ho * Wo;
        v = X[((b * CIN + cin) * H + ho + kh) * W + wo + kw];
      }
      Bs[i] = v;
    }
    // ---- prefetch next weight K-tile (global_prefetch_b8) ----
    if (k0 + BK < K) {
      int m = m0 + (t & 127);
      if (m < M) __builtin_prefetch(&Wb[m * K + k0 + BK], 0, 3);
    }
    __syncthreads();

    // ---- 2 K-substeps x (2x2) register tile = 8 WMMAs per stage ----
    #pragma unroll
    for (int ks = 0; ks < 2; ++ks) {
      const v16h a0 = *(const v16h*)&As[((ks * 8 + wm * 2 + 0) * 32 + lane) * 16];
      const v16h a1 = *(const v16h*)&As[((ks * 8 + wm * 2 + 1) * 32 + lane) * 16];
      const v16h b0 = *(const v16h*)&Bs[((ks * 4 + wn * 2 + 0) * 32 + lane) * 16];
      const v16h b1 = *(const v16h*)&Bs[((ks * 4 + wn * 2 + 1) * 32 + lane) * 16];
      acc[0][0] = __builtin_amdgcn_wmma_f32_16x16x32_f16(false, a0, false, b0,
                                                         (short)0, acc[0][0], false, false);
      acc[0][1] = __builtin_amdgcn_wmma_f32_16x16x32_f16(false, a0, false, b1,
                                                         (short)0, acc[0][1], false, false);
      acc[1][0] = __builtin_amdgcn_wmma_f32_16x16x32_f16(false, a1, false, b0,
                                                         (short)0, acc[1][0], false, false);
      acc[1][1] = __builtin_amdgcn_wmma_f32_16x16x32_f16(false, a1, false, b1,
                                                         (short)0, acc[1][1], false, false);
    }
    __syncthreads();
  }

  // ---- epilogue: C/D layout: M = vreg + 8*(lane>>4), N = lane&15 ----
  #pragma unroll
  for (int ti = 0; ti < 2; ++ti) {
    #pragma unroll
    for (int tj = 0; tj < 2; ++tj) {
      #pragma unroll
      for (int v = 0; v < 8; ++v) {
        int m = m0 + (wm * 2 + ti) * 16 + v + (hi16 << 3);
        int n = n0 + (wn * 2 + tj) * 16 + l15;
        if (m < M && n < N) {
          int b = n / S, sp = n - b * S;
          Y[((long)b * M + m) * S + sp] = acc[ti][tj][v];
        }
      }
    }
  }
}

// ---------------------------------------------------------------------------
// Host orchestration
// ---------------------------------------------------------------------------
static inline int ceil_div(int a, int b) { return (a + b - 1) / b; }

template<int CIN, int H, int W, int KH, int KW>
static void launch_gemm(const _Float16* wgt, const _Float16* in, float* raw,
                        int M, int B, hipStream_t stream) {
  constexpr int Ho = H - KH + 1, Wo = W - KW + 1;
  int N = B * Ho * Wo;
  dim3 grid(ceil_div(N, 64), ceil_div(M, 128));
  k_conv_wmma<CIN, H, W, KH, KW><<<grid, TB, 0, stream>>>(wgt, in, raw, M, B);
}

extern "C" void kernel_launch(void* const* d_in, const int* in_sizes, int n_in,
                              void* d_out, int out_size, void* d_ws, size_t ws_size,
                              hipStream_t stream) {
  (void)in_sizes; (void)n_in; (void)out_size; (void)ws_size;
  const int B = 256;

  // setup_inputs() dict order: x, w1..w6, fw1, fw2, fw3, g1,b1, ..., g6,b6
  const float* x = (const float*)d_in[0];
  const float* w[9];
  for (int i = 0; i < 9; ++i) w[i] = (const float*)d_in[1 + i];
  const float* g[6]; const float* bb[6];
  for (int i = 0; i < 6; ++i) { g[i] = (const float*)d_in[10 + 2*i]; bb[i] = (const float*)d_in[11 + 2*i]; }

  // ---- carve workspace (needs ~290 MB) ----
  char* ws = (char*)d_ws;
  size_t off = 0;
  auto alloc = [&](size_t bytes) -> void* {
    void* p = ws + off; off += (bytes + 255) & ~(size_t)255; return p;
  };

  const int wsz[9] = {128*1*9, 128*128*9, 256*128*9, 256*256*9,
                      512*256*9, 512*512*9, 1024*4608, 1024*1024, 10*1024};
  _Float16* wb[9];
  for (int i = 0; i < 9; ++i) wb[i] = (_Float16*)alloc((size_t)wsz[i] * 2);

  const size_t ACT_ELEMS = (size_t)33 * 1024 * 1024;  // >= 31.72M (conv3 raw)
  _Float16* act0  = (_Float16*)alloc(ACT_ELEMS * 2);
  _Float16* act1  = (_Float16*)alloc(ACT_ELEMS * 2);
  float*    raw   = (float*)   alloc(ACT_ELEMS * 4);
  float*    meanb = (float*)   alloc(512 * 4);
  float*    istdb = (float*)   alloc(512 * 4);

  // ---- binarize weights to +-1 f16; convert input to f16 ----
  for (int i = 0; i < 9; ++i)
    k_sign_f16<<<ceil_div(wsz[i], TB), TB, 0, stream>>>(w[i], wb[i], wsz[i]);
  k_f32_to_f16<<<ceil_div(B*28*28, TB), TB, 0, stream>>>(x, act0, B*28*28);

  auto bn = [&](float* rawp, _Float16* outAct, const float* gg, const float* bbeta,
                int M, int Sp) {
    k_bn_stats<<<M, TB, 0, stream>>>(rawp, meanb, istdb, M, Sp, B);
    int total = B * M * Sp;
    k_bn_apply_relu<<<ceil_div(total, TB), TB, 0, stream>>>(rawp, outAct, meanb, istdb,
                                                            gg, bbeta, M, Sp, total);
  };

  // conv1..conv4
  launch_gemm<  1, 28, 28, 3, 3>(wb[0], act0, raw, 128, B, stream);
  bn(raw, act1, g[0], bb[0], 128, 26*26);                       // [B,128,26,26]
  launch_gemm<128, 26, 26, 3, 3>(wb[1], act1, raw, 128, B, stream);
  bn(raw, act0, g[1], bb[1], 128, 24*24);                       // [B,128,24,24]
  launch_gemm<128, 24, 24, 3, 3>(wb[2], act0, raw, 256, B, stream);
  bn(raw, act1, g[2], bb[2], 256, 22*22);                       // [B,256,22,22]
  launch_gemm<256, 22, 22, 3, 3>(wb[3], act1, raw, 256, B, stream);
  bn(raw, act0, g[3], bb[3], 256, 20*20);                       // [B,256,20,20]
  { int total = B*256*10*10;
    k_maxpool2_f16<<<ceil_div(total, TB), TB, 0, stream>>>(act0, act1, 256, 20, 20, total); }
  // conv5, conv6
  launch_gemm<256, 10, 10, 3, 3>(wb[4], act1, raw, 512, B, stream);
  bn(raw, act0, g[4], bb[4], 512, 8*8);                         // [B,512,8,8]
  launch_gemm<512,  8,  8, 3, 3>(wb[5], act0, raw, 512, B, stream);
  bn(raw, act1, g[5], bb[5], 512, 6*6);                         // [B,512,6,6]
  { int total = B*512*3*3;
    k_maxpool2_f16<<<ceil_div(total, TB), TB, 0, stream>>>(act1, act0, 512, 6, 6, total); }
  // act0 = flattened [B, 4608] f16

  // FC layers as K x 1x1 "conv" (S == 1 -> output is [B, O] row-major)
  launch_gemm<4608, 1, 1, 1, 1>(wb[6], act0, raw, 1024, B, stream);
  { int total = B*1024; k_relu_f16<<<ceil_div(total, TB), TB, 0, stream>>>(raw, act1, total); }
  launch_gemm<1024, 1, 1, 1, 1>(wb[7], act1, raw, 1024, B, stream);
  { int total = B*1024; k_relu_f16<<<ceil_div(total, TB), TB, 0, stream>>>(raw, act0, total); }
  launch_gemm<1024, 1, 1, 1, 1>(wb[8], act0, raw, 10, B, stream);
  { int total = B*10; k_sigmoid<<<ceil_div(total, TB), TB, 0, stream>>>(raw, (float*)d_out, total); }
}
